// past_decoder_8366596293117
// MI455X (gfx1250) — compile-verified, matrix operand order checked by reading, not run on hardware
//
#include <hip/hip_runtime.h>

#define T_STEPS 20
#define NAG 131072
#define HID 192
#define WAVES_PER_WG 4
#define ROWS_PER_WG 64
#define NBT (48 * 6) // 288 B-fragment tiles of w_hh^T

typedef __attribute__((ext_vector_type(16))) _Float16 v16h;
typedef __attribute__((ext_vector_type(8)))  _Float16 v8h;
typedef __attribute__((ext_vector_type(2)))  _Float16 v2h;
typedef __attribute__((ext_vector_type(8)))  float    v8f;
typedef __attribute__((ext_vector_type(2)))  float    v2f;

__device__ __forceinline__ float fast_exp(float x) {
    return __builtin_amdgcn_exp2f(x * 1.44269504088896340736f);
}
__device__ __forceinline__ float fast_sigmoid(float x) {
    return __builtin_amdgcn_rcpf(1.0f + fast_exp(-x));
}
__device__ __forceinline__ float fast_tanh(float x) {
    return 1.0f - 2.0f * __builtin_amdgcn_rcpf(fast_exp(2.0f * x) + 1.0f);
}

__global__ __launch_bounds__(128, 1)
void lstm_decoder_persistent(const float* __restrict__ obs,   // (20, N, 2)
                             const float* __restrict__ h0,    // (N, 192)
                             const float* __restrict__ w_ih,  // (768, 2)
                             const float* __restrict__ w_hh,  // (768, 192)
                             const float* __restrict__ b_ih,  // (768)
                             const float* __restrict__ b_hh,  // (768)
                             const float* __restrict__ w_out, // (2, 192)
                             const float* __restrict__ b_out, // (2)
                             float* __restrict__ out)         // (20, N, 2)
{
    // 294912 B: w_hh^T pre-swizzled into per-lane f16 WMMA B fragments
    __shared__ __align__(32) _Float16 sB[NBT * 32 * 16];
    // 24576 B: per-wave h staging (row-major f16) for D-layout -> A-layout transpose
    __shared__ __align__(16) _Float16 sH[WAVES_PER_WG][16][HID];
    // 6144 B: pre-swizzled B fragments for the x/bias f32 WMMA (16x16x4).
    // Per tile ct (gate col group), lane ln: ln<16 -> {w_ih[j,0], w_ih[j,1]},
    // ln>=16 -> {b_ih[j]+b_hh[j], 0}, j = ct*16 + (ln&15). (K rows 0,1 on low
    // lane half; K rows 2,3 on high half, mirroring the 32-bit 16x4 A layout.)
    __shared__ __align__(8) _Float16 sXB[48 * 32 * 2];
    __shared__ float sWout[2 * HID]; // 1536 B

    const int tid  = threadIdx.x;
    const int lane = tid & 31;
    const int wv   = tid >> 5;
    const int hi   = lane >> 4;   // lane half
    const int ln16 = lane & 15;
    const int aBase = (blockIdx.x * WAVES_PER_WG + wv) * 16;

    // ---- one-time cooperative loads / swizzles ----
    // f16 B fragment element i (0..15) of lane ln in tile (ct,kt) holds
    // K = kt*32 + (i<8 ? 8*half+i : 16 + 8*half + (i-8)), N = ct*16 + (ln&15)
    for (int e = tid; e < NBT * 32 * 16; e += 128) {
        int i    = e & 15;
        int ln   = (e >> 4) & 31;
        int tile = e >> 9;
        int ct = tile / 6, kt = tile - ct * 6;
        int lh = ln >> 4, n = ln & 15;
        int kl = (i < 8) ? (lh * 8 + i) : (8 + lh * 8 + i);
        sB[e] = (_Float16)w_hh[(ct * 16 + n) * HID + (kt * 32 + kl)];
    }
    for (int e = tid; e < 48 * 32 * 2; e += 128) {
        int pos = e & 1;
        int ln  = (e >> 1) & 31;
        int ct  = e >> 6;
        int j   = ct * 16 + (ln & 15);
        float v;
        if (ln < 16) v = w_ih[2 * j + pos];
        else         v = pos ? 0.0f : (b_ih[j] + b_hh[j]);
        sXB[e] = (_Float16)v;
    }
    for (int e = tid; e < 2 * HID; e += 128) sWout[e] = w_out[e];
    // initial hidden state into staging (f16)
    for (int e = lane; e < 16 * HID; e += 32) {
        int r = e / HID, c = e - r * HID;
        sH[wv][r][c] = (_Float16)h0[(size_t)(aBase + r) * HID + c];
    }
    __syncthreads();

    const float bo0 = b_out[0], bo1 = b_out[1];
    const float2* obs2 = (const float2*)obs;
    const v8f kZero = {};

    v8f cfr[12]; // cell state, D-layout, 192 cols per row tile
    #pragma unroll
    for (int i = 0; i < 12; ++i) cfr[i] = kZero;

    for (int t = 0; t < T_STEPS; ++t) {
        // ---- rebuild A fragments of h from per-wave staging ----
        asm volatile("s_wait_dscnt 0" ::: "memory");
        v16h a[6];
        #pragma unroll
        for (int kt = 0; kt < 6; ++kt) {
            const _Float16* p = &sH[wv][ln16][kt * 32 + hi * 8];
            v8h lo = *(const v8h*)p;
            v8h hh = *(const v8h*)(p + 16);
            a[kt] = __builtin_shufflevector(lo, hh,
                     0,1,2,3,4,5,6,7,8,9,10,11,12,13,14,15);
        }
        // ---- A fragment for the x/bias f32 WMMA: [x0, x1, 1, 0] per row ----
        const int srcT = (t == 0) ? 0 : (t - 1);
        const float2* xrow = obs2 + (size_t)srcT * NAG;
        v2f ax;
        if (hi == 0) {
            float2 x = xrow[aBase + ln16];
            ax[0] = x.x; ax[1] = x.y;        // K = 0,1
        } else {
            ax[0] = 1.0f; ax[1] = 0.0f;      // K = 2 (bias row), 3 (zero)
        }

        float o0[8], o1[8];
        #pragma unroll
        for (int r = 0; r < 8; ++r) { o0[r] = 0.0f; o1[r] = 0.0f; }

        #pragma unroll
        for (int cg = 0; cg < 12; ++cg) {
            const int col = cg * 16 + ln16;
            v8f acc[4];
            #pragma unroll
            for (int g = 0; g < 4; ++g) {
                const int ct = g * 12 + cg;
                // x*w_ih^T + bias via 16x16x4 f32 WMMA (C = 0)
                v2h xb2 = *(const v2h*)&sXB[(ct * 32 + lane) * 2];
                v2f bx; bx[0] = (float)xb2[0]; bx[1] = (float)xb2[1];
                acc[g] = __builtin_amdgcn_wmma_f32_16x16x4_f32(
                    false, ax, false, bx, (short)0, kZero, false, false);
                // h @ w_hh^T accumulated on top
                #pragma unroll
                for (int kt = 0; kt < 6; ++kt) {
                    const v16h b = *(const v16h*)&sB[(size_t)((ct * 6 + kt) * 32 + lane) * 16];
                    acc[g] = __builtin_amdgcn_wmma_f32_16x16x32_f16(
                        false, a[kt], false, b, (short)0, acc[g], false, false);
                }
            }
            // ---- LSTM cell elementwise for these 16 hidden columns ----
            v8f c = cfr[cg];
            const float wo0 = sWout[col], wo1 = sWout[HID + col];
            #pragma unroll
            for (int r = 0; r < 8; ++r) {
                float ig = fast_sigmoid(acc[0][r]);
                float fg = fast_sigmoid(acc[1][r]);
                float gg = fast_tanh(acc[2][r]);
                float og = fast_sigmoid(acc[3][r]);
                float cn = fg * c[r] + ig * gg;
                c[r] = cn;
                float hn = og * fast_tanh(cn);
                o0[r] += hn * wo0;
                o1[r] += hn * wo1;
                sH[wv][r + 8 * hi][col] = (_Float16)hn; // restage for next step
            }
            cfr[cg] = c;
        }
        // ---- output projection: reduce across the 16-lane column group ----
        #pragma unroll
        for (int m = 8; m >= 1; m >>= 1) {
            #pragma unroll
            for (int r = 0; r < 8; ++r) {
                o0[r] += __shfl_xor(o0[r], m, 32);
                o1[r] += __shfl_xor(o1[r], m, 32);
            }
        }
        if (ln16 == 0) {
            float2* op = (float2*)out + (size_t)t * NAG + aBase + 8 * hi;
            #pragma unroll
            for (int r = 0; r < 8; ++r) {
                float2 v; v.x = o0[r] + bo0; v.y = o1[r] + bo1;
                op[r] = v;
            }
        }
    }
}

extern "C" void kernel_launch(void* const* d_in, const int* in_sizes, int n_in,
                              void* d_out, int out_size, void* d_ws, size_t ws_size,
                              hipStream_t stream) {
    const float* obs   = (const float*)d_in[0];
    const float* h0    = (const float*)d_in[1];
    const float* w_ih  = (const float*)d_in[2];
    const float* w_hh  = (const float*)d_in[3];
    const float* b_ih  = (const float*)d_in[4];
    const float* b_hh  = (const float*)d_in[5];
    const float* w_out = (const float*)d_in[6];
    const float* b_out = (const float*)d_in[7];
    float* out = (float*)d_out;

    (void)in_sizes; (void)n_in; (void)out_size; (void)d_ws; (void)ws_size;

    lstm_decoder_persistent<<<NAG / ROWS_PER_WG, 128, 0, stream>>>(
        obs, h0, w_ih, w_hh, b_ih, b_hh, w_out, b_out, out);
}
